// AttentionLayer_89953795047745
// MI455X (gfx1250) — matrix operand order repaired and checked by reading, not compiled, not run on hardware
//
#include <hip/hip_runtime.h>

typedef __attribute__((ext_vector_type(16))) __bf16 v16bf;
typedef __attribute__((ext_vector_type(8)))  float  v8f;
typedef __attribute__((ext_vector_type(4)))  int    v4i;

#define C_DIM 1024
#define T_DIM 8192
#define B_DIM 2
#define H_DIM 16
#define D_DIM 64
#define W_WIN 512
#define NROWS (B_DIM * T_DIM)   // 16384

// ---------- gfx1250 async global->LDS (ASYNCcnt-tracked DMA) with fallback ----------
#if defined(__gfx1250__) && __has_builtin(__builtin_amdgcn_global_load_async_to_lds_b128)
#define HAVE_ASYNC_LDS 1
#else
#define HAVE_ASYNC_LDS 0
#endif

#if HAVE_ASYNC_LDS
typedef __attribute__((address_space(1))) v4i* gv4i_ptr;
typedef __attribute__((address_space(3))) v4i* lv4i_ptr;
#endif

__device__ __forceinline__ void async_copy16(const __bf16* g, __bf16* l) {
#if HAVE_ASYNC_LDS
  __builtin_amdgcn_global_load_async_to_lds_b128(
      (gv4i_ptr)(void*)g, (lv4i_ptr)(void*)l, 0, 0);
  __builtin_amdgcn_global_load_async_to_lds_b128(
      (gv4i_ptr)(void*)(g + 8), (lv4i_ptr)(void*)(l + 8), 0, 0);
#else
  uint4 t0 = *(const uint4*)g;
  uint4 t1 = *(const uint4*)(g + 8);
  *(uint4*)l = t0;
  *(uint4*)(l + 8) = t1;
#endif
}

__device__ __forceinline__ void wait_async_all() {
#if HAVE_ASYNC_LDS
#if __has_builtin(__builtin_amdgcn_s_wait_asynccnt)
  __builtin_amdgcn_s_wait_asynccnt(0);
#else
  asm volatile("s_wait_asynccnt 0x0" ::: "memory");
#endif
#endif
}

// ---------- helpers ----------
__device__ __forceinline__ v16bf load16(const __bf16* p) {
  union { v16bf v; uint4 q[2]; } u;
  u.q[0] = *(const uint4*)(p);
  u.q[1] = *(const uint4*)(p + 8);
  return u.v;
}

// ---------- fp32 -> bf16 converts ----------
__global__ void cvt_x_kernel(const float* __restrict__ x, __bf16* __restrict__ y, int n) {
  int i = blockIdx.x * blockDim.x + threadIdx.x;
  if (i < n) y[i] = (__bf16)x[i];
}

// wT[k][m] = W[m][k]  (so GEMM B-tiles are K-major contiguous)
__global__ void cvt_wT_kernel(const float* __restrict__ w, __bf16* __restrict__ wt) {
  int i = blockIdx.x * blockDim.x + threadIdx.x;   // 1M elements
  if (i >= C_DIM * C_DIM) return;
  int m = i >> 10, k = i & (C_DIM - 1);
  wt[(size_t)k * C_DIM + m] = (__bf16)w[i];
}

// ---------- bf16 GEMM: out[m_sample, n_chan] = sum_k A[m,k] * Wt[k,n] ----------
// MODE 0: bf16 out [M, C];  MODE 1: bf16 out transposed per-head [B,H,D,T];  MODE 2: fp32 out [M, C]
template <int MODE>
__global__ __launch_bounds__(128)
void gemm_bf16_kernel(const __bf16* __restrict__ A, const __bf16* __restrict__ Bt,
                      void* __restrict__ Out)
{
  const int M0 = blockIdx.y * 64;
  const int N0 = blockIdx.x * 64;
  const int tid  = threadIdx.x;
  const int lane = tid & 31;
  const int wv   = tid >> 5;
  const int hi   = lane >> 4;
  const int l16  = lane & 15;

  __shared__ __bf16 Atile[2][64 * 40];   // stride 40 halfs: 16B-aligned rows, bank-spread
  __shared__ __bf16 Btile[2][32 * 72];   // stride 72 halfs

  // per-thread staging coordinates (fixed across k-slabs)
  const int rowA = tid >> 1, segA = tid & 1;    // 64 rows x 2 segs of 16 halfs
  const int rowB = tid >> 2, segB = tid & 3;    // 32 rows x 4 segs of 16 halfs
  const __bf16* srcA = A  + (size_t)(M0 + rowA) * C_DIM + segA * 16;
  const __bf16* srcB = Bt + (size_t)rowB * C_DIM + N0 + segB * 16;
  const int dstAOff = rowA * 40 + segA * 16;
  const int dstBOff = rowB * 72 + segB * 16;

  v8f acc[2][2] = {};

  // prologue: stage slab 0 into buffer 0
  async_copy16(srcA, &Atile[0][dstAOff]);
  async_copy16(srcB, &Btile[0][dstBOff]);
  wait_async_all();
  __syncthreads();

  const int NIT = C_DIM / 32;
  for (int it = 0; it < NIT; ++it) {
    const int cur = it & 1;
    // kick off DMA for next slab into the other buffer (overlaps with WMMA below)
    if (it + 1 < NIT) {
      const int kt = (it + 1) * 32;
      async_copy16(srcA + kt,                   &Atile[1 - cur][dstAOff]);
      async_copy16(srcB + (size_t)kt * C_DIM,   &Btile[1 - cur][dstBOff]);
    }

    // fragments from current buffer
    v16bf a[2], b[2];
    #pragma unroll
    for (int mi = 0; mi < 2; ++mi) {
      const __bf16* rp = &Atile[cur][((wv & 1) * 32 + mi * 16 + l16) * 40];
      union { v16bf v; unsigned int u[8]; } ua;
      #pragma unroll
      for (int v = 0; v < 8; ++v) {
        int k = ((v < 4) ? 2 * v : 16 + 2 * (v - 4)) + 8 * hi;   // ISA A-frag K pattern
        ua.u[v] = *(const unsigned int*)(rp + k);
      }
      a[mi] = ua.v;
    }
    #pragma unroll
    for (int ni = 0; ni < 2; ++ni) {
      union { v16bf v; __bf16 e[16]; } ub;
      #pragma unroll
      for (int s = 0; s < 16; ++s) {
        int k = ((s < 8) ? s : 8 + s) + 8 * hi;
        ub.e[s] = Btile[cur][k * 72 + (wv >> 1) * 32 + ni * 16 + l16];
      }
      b[ni] = ub.v;
    }

    #pragma unroll
    for (int mi = 0; mi < 2; ++mi)
      #pragma unroll
      for (int ni = 0; ni < 2; ++ni)
        acc[mi][ni] = __builtin_amdgcn_wmma_f32_16x16x32_bf16(
            false, a[mi], false, b[ni], (short)0, acc[mi][ni], false, false);

    wait_async_all();
    __syncthreads();
  }

  // branch-free epilogue (MODE is compile-time)
  const int mBase = M0 + (wv & 1) * 32;
  const int nBase = N0 + (wv >> 1) * 32;
  #pragma unroll
  for (int mi = 0; mi < 2; ++mi) {
    #pragma unroll
    for (int ni = 0; ni < 2; ++ni) {
      int ncol = nBase + ni * 16 + l16;
      #pragma unroll
      for (int v = 0; v < 8; ++v) {
        int mrow = mBase + mi * 16 + v + 8 * hi;
        float val = acc[mi][ni][v];
        if constexpr (MODE == 2) {
          ((float*)Out)[(size_t)mrow * C_DIM + ncol] = val;
        } else if constexpr (MODE == 0) {
          ((__bf16*)Out)[(size_t)mrow * C_DIM + ncol] = (__bf16)val;
        } else { // kT layout [B,H,D,T]
          int bb = mrow >> 13;      // / T_DIM
          int tt = mrow & (T_DIM - 1);
          int hh = ncol >> 6;
          int dd = ncol & (D_DIM - 1);
          ((__bf16*)Out)[(((size_t)bb * H_DIM + hh) * D_DIM + dd) * T_DIM + tt] = (__bf16)val;
        }
      }
    }
  }
}

// ---------- sliding-window attention (flash-style, bf16 WMMA, fp32 softmax) ----------
__global__ __launch_bounds__(128)
void attn_kernel(const __bf16* __restrict__ Q,    // [B*T, C]
                 const __bf16* __restrict__ Kt,   // [B,H,D,T]
                 const __bf16* __restrict__ V,    // [B*T, C]
                 __bf16* __restrict__ O)          // [B*T, C]
{
  const int qtile = blockIdx.x;        // 0..7   (64 queries each)
  const int nblk  = blockIdx.y;        // 0..15
  const int bh    = blockIdx.z;        // 0..31
  const int b = bh >> 4, h = bh & 15;
  const int tid = threadIdx.x;
  const int lane = tid & 31, wv = tid >> 5;
  const int hi = lane >> 4, l16 = lane & 15;

  const int i0 = qtile * 64;           // block-local query tile start
  const int iw = i0 + wv * 16;         // this wave's 16 query rows
  const float scale = 0.125f;          // 1/sqrt(D)
  const float L2E = 1.44269504088896f;

  __shared__ __bf16 Pbuf[4][16 * 40];  // per-wave P staging (C/D layout -> A layout)

  // Q A-fragments, loaded once and reused over all key chunks
  v16bf qa[2];
  {
    size_t row = (size_t)(b * T_DIM) + (size_t)nblk * W_WIN + (iw + l16);
    const __bf16* qp = Q + row * C_DIM + h * D_DIM;
    #pragma unroll
    for (int f = 0; f < 2; ++f) {
      union { v16bf v; unsigned int u[8]; } ua;
      #pragma unroll
      for (int v = 0; v < 8; ++v) {
        int d = f * 32 + ((v < 4) ? 2 * v : 16 + 2 * (v - 4)) + 8 * hi;
        ua.u[v] = *(const unsigned int*)(qp + d);
      }
      qa[f] = ua.v;
    }
  }

  v8f oacc[4] = {};
  float mrow[8], lrow[8];
  #pragma unroll
  for (int v = 0; v < 8; ++v) { mrow[v] = -1e30f; lrow[v] = 0.f; }

  const __bf16* ktBase = Kt + (size_t)bh * D_DIM * T_DIM;
  const int jstart = (nblk == 0) ? W_WIN : i0;
  const int jend   = i0 + W_WIN + 64;             // exclusive, <= 2W

  for (int j0 = jstart; j0 < jend; j0 += 32) {
    const int tk0 = (nblk - 1) * W_WIN + j0;      // global key time base (>=0)

    // prefetch next chunk's K/V streams into cache (global_prefetch_b8)
    if (j0 + 32 < jend) {
      const int tkn = tk0 + 32;
      __builtin_prefetch(ktBase + (size_t)lane * T_DIM + tkn, 0, 3);
      __builtin_prefetch(V + ((size_t)(b * T_DIM) + tkn + lane) * C_DIM + h * D_DIM, 0, 3);
    }

    // ---- scores: S[16 x 32] as two C/D frags, K-contraction over D=64 ----
    v8f sf[2] = {};
    #pragma unroll
    for (int kh = 0; kh < 2; ++kh) {
      #pragma unroll
      for (int f = 0; f < 2; ++f) {
        const __bf16* kp = ktBase + tk0 + kh * 16 + l16;
        union { v16bf v; __bf16 e[16]; } ubk;
        #pragma unroll
        for (int s = 0; s < 16; ++s) {
          int d = f * 32 + ((s < 8) ? s : 8 + s) + 8 * hi;
          ubk.e[s] = kp[(size_t)d * T_DIM];
        }
        v16bf bk = ubk.v;
        sf[kh] = __builtin_amdgcn_wmma_f32_16x16x32_bf16(
            false, qa[f], false, bk, (short)0, sf[kh], false, false);
      }
    }

    // ---- mask + scale + chunk row-max ----
    float cmax[8];
    #pragma unroll
    for (int v = 0; v < 8; ++v) cmax[v] = -1e30f;
    #pragma unroll
    for (int kh = 0; kh < 2; ++kh) {
      #pragma unroll
      for (int v = 0; v < 8; ++v) {
        int irow = iw + v + 8 * hi;
        int j = j0 + kh * 16 + l16;
        bool ok = (j > irow) && (j <= irow + W_WIN);
        float s = ok ? sf[kh][v] * scale : -1e30f;
        sf[kh][v] = s;
        cmax[v] = fmaxf(cmax[v], s);
      }
    }
    #pragma unroll
    for (int v = 0; v < 8; ++v) {
      float x = cmax[v];
      x = fmaxf(x, __shfl_xor(x, 1, 16));
      x = fmaxf(x, __shfl_xor(x, 2, 16));
      x = fmaxf(x, __shfl_xor(x, 4, 16));
      x = fmaxf(x, __shfl_xor(x, 8, 16));
      cmax[v] = x;
    }

    // ---- online softmax update ----
    float corr[8], rsum[8];
    #pragma unroll
    for (int v = 0; v < 8; ++v) {
      float mnew = fmaxf(mrow[v], cmax[v]);
      corr[v] = exp2f((mrow[v] - mnew) * L2E);
      mrow[v] = mnew;
      rsum[v] = 0.f;
    }

    __bf16* pb = Pbuf[wv];
    #pragma unroll
    for (int kh = 0; kh < 2; ++kh) {
      #pragma unroll
      for (int v = 0; v < 8; ++v) {
        float s = sf[kh][v];
        float p = (s > -0.5e30f) ? exp2f((s - mrow[v]) * L2E) : 0.f;
        rsum[v] += p;
        pb[(v + 8 * hi) * 40 + kh * 16 + l16] = (__bf16)p;
      }
    }
    #pragma unroll
    for (int v = 0; v < 8; ++v) {
      float x = rsum[v];
      x += __shfl_xor(x, 1, 16);
      x += __shfl_xor(x, 2, 16);
      x += __shfl_xor(x, 4, 16);
      x += __shfl_xor(x, 8, 16);
      lrow[v] = lrow[v] * corr[v] + x;
    }
    #pragma unroll
    for (int f = 0; f < 4; ++f)
      #pragma unroll
      for (int v = 0; v < 8; ++v)
        oacc[f][v] *= corr[v];

    // ---- P (A-frag from LDS) x V ----
    union { v16bf v; unsigned int u[8]; } up;
    {
      const __bf16* rp = &pb[l16 * 40];
      #pragma unroll
      for (int v = 0; v < 8; ++v) {
        int k = ((v < 4) ? 2 * v : 16 + 2 * (v - 4)) + 8 * hi;
        up.u[v] = *(const unsigned int*)(rp + k);
      }
    }
    const __bf16* vrow = V + ((size_t)(b * T_DIM) + tk0) * C_DIM + h * D_DIM + l16;
    #pragma unroll
    for (int f = 0; f < 4; ++f) {
      union { v16bf v; __bf16 e[16]; } ubv;
      #pragma unroll
      for (int s = 0; s < 16; ++s) {
        int kk = ((s < 8) ? s : 8 + s) + 8 * hi;
        ubv.e[s] = vrow[(size_t)kk * C_DIM + f * 16];
      }
      v16bf bv = ubv.v;
      oacc[f] = __builtin_amdgcn_wmma_f32_16x16x32_bf16(
          false, up.v, false, bv, (short)0, oacc[f], false, false);
    }
  }

  // ---- normalize + store bf16 ----
  const size_t rowBase = (size_t)(b * T_DIM) + (size_t)nblk * W_WIN;
  #pragma unroll
  for (int v = 0; v < 8; ++v) {
    float inv = 1.0f / lrow[v];
    int irow = iw + v + 8 * hi;
    __bf16* op = O + (rowBase + irow) * C_DIM + h * D_DIM + l16;
    #pragma unroll
    for (int f = 0; f < 4; ++f)
      op[f * 16] = (__bf16)(oacc[f][v] * inv);
  }
}

// ---------- host ----------
extern "C" void kernel_launch(void* const* d_in, const int* in_sizes, int n_in,
                              void* d_out, int out_size, void* d_ws, size_t ws_size,
                              hipStream_t stream)
{
  (void)in_sizes; (void)n_in; (void)out_size; (void)ws_size;
  const float* x  = (const float*)d_in[0];
  const float* Wq = (const float*)d_in[1];
  const float* Wk = (const float*)d_in[2];
  const float* Wv = (const float*)d_in[3];
  const float* Wo = (const float*)d_in[4];

  char* ws = (char*)d_ws;
  size_t off = 0;
  auto alloc = [&](size_t bytes) {
    char* p = ws + off;
    off += (bytes + 255) & ~(size_t)255;
    return p;
  };
  const size_t actBytes = (size_t)NROWS * C_DIM * 2;
  const size_t wBytes   = (size_t)C_DIM * C_DIM * 2;
  __bf16* xh    = (__bf16*)alloc(actBytes);
  __bf16* wqT   = (__bf16*)alloc(wBytes);
  __bf16* wkT   = (__bf16*)alloc(wBytes);
  __bf16* wvT   = (__bf16*)alloc(wBytes);
  __bf16* woT   = (__bf16*)alloc(wBytes);
  __bf16* qbf   = (__bf16*)alloc(actBytes);
  __bf16* kT    = (__bf16*)alloc(actBytes);
  __bf16* vbf   = (__bf16*)alloc(actBytes);
  __bf16* attbf = (__bf16*)alloc(actBytes);

  const int nx = NROWS * C_DIM;
  cvt_x_kernel<<<(nx + 255) / 256, 256, 0, stream>>>(x, xh, nx);
  const int nw = C_DIM * C_DIM;
  cvt_wT_kernel<<<(nw + 255) / 256, 256, 0, stream>>>(Wq, wqT);
  cvt_wT_kernel<<<(nw + 255) / 256, 256, 0, stream>>>(Wk, wkT);
  cvt_wT_kernel<<<(nw + 255) / 256, 256, 0, stream>>>(Wv, wvT);
  cvt_wT_kernel<<<(nw + 255) / 256, 256, 0, stream>>>(Wo, woT);

  dim3 gg(C_DIM / 64, NROWS / 64);
  gemm_bf16_kernel<0><<<gg, 128, 0, stream>>>(xh, wqT, qbf);
  gemm_bf16_kernel<1><<<gg, 128, 0, stream>>>(xh, wkT, kT);   // K pre-transposed [B,H,D,T]
  gemm_bf16_kernel<0><<<gg, 128, 0, stream>>>(xh, wvT, vbf);

  dim3 ga(W_WIN / 64, T_DIM / W_WIN, B_DIM * H_DIM);
  attn_kernel<<<ga, 128, 0, stream>>>(qbf, kT, vbf, attbf);

  gemm_bf16_kernel<2><<<gg, 128, 0, stream>>>(attbf, woT, d_out);
}
